// DynamicVarsDecoder_87454124081345
// MI455X (gfx1250) — matrix operand order repaired and checked by reading, not compiled, not run on hardware
//
#include <hip/hip_runtime.h>
#include <cstdint>

typedef unsigned int u32;
typedef __attribute__((ext_vector_type(16))) _Float16 v16h;
typedef __attribute__((ext_vector_type(8)))  float    v8f;

#define WAVES 4

// Fast silu: sigmoid via v_rcp_f32 instead of IEEE-correct division.
__device__ __forceinline__ float silu_f(float x) {
  float e = __expf(-x);
  return x * __builtin_amdgcn_rcpf(1.0f + e);
}

__device__ __forceinline__ u32 pack2h(float a, float b) {
  _Float16 ha = (_Float16)a, hb = (_Float16)b;
  unsigned short ua = __builtin_bit_cast(unsigned short, ha);
  unsigned short ub = __builtin_bit_cast(unsigned short, hb);
  return (u32)ua | ((u32)ub << 16);
}

// Wave-private LDS producer->consumer ordering: DS ops execute in-order at the
// LDS within a wave (ISA ch.11); we only need to stop compiler code motion.
__device__ __forceinline__ void wave_fence() {
  __builtin_amdgcn_wave_barrier();
  asm volatile("" ::: "memory");
}

// A fragment (16x32 f16, ISA 7.12.2): lane holds row = lane&15; halves split K.
// xp = dword view of row-major f16 [16][2*stridePairs]; chunk c covers K=32c..32c+31.
__device__ __forceinline__ v16h load_afrag(const u32* xp, int stridePairs, int c, int lane) {
  const int row = lane & 15, hf = lane >> 4;
  union { u32 u[8]; v16h v; } r;
#pragma unroll
  for (int j = 0; j < 8; ++j) {
    const int pr = j + 8 * hf;  // K-pair in chunk
    r.u[j] = xp[row * stridePairs + c * 16 + pr];
  }
  return r.v;
}

// B fragment from fragment-major swizzled weights: [frag][lane][8 dwords].
// One contiguous 32B per lane -> 2x ds_load_b128 straight into operand regs.
__device__ __forceinline__ v16h load_bfrag_swz(const u32* wsz, int f, int lane) {
  return ((const v16h*)wsz)[f * 32 + lane];
}

#define WMMA_F16(a, b, c) \
  __builtin_amdgcn_wmma_f32_16x16x32_f16(false, (a), false, (b), (short)0, (c), false, false)

// Swizzle a [2K x 64] f32 weight matrix into fragment-major packed-f16 LDS:
// frag f = c*4 + tt ; within frag: lane, j ; element = (W[2*pr][col], W[2*pr+1][col])
// with pr = 16c + 8*(lane>>4) + j, col = 16*tt + (lane&15).
__device__ __forceinline__ void swz_idx(int i, int& pr, int& col, int& f) {
  f = i >> 8;
  int ln = (i >> 3) & 31, j = i & 7;
  int c = f >> 2, tt = f & 3;
  pr = 16 * c + 8 * (ln >> 4) + j;
  col = 16 * tt + (ln & 15);
}

// ---------------------------------------------------------------- helpers ---

__global__ void egnn_zero(float* __restrict__ aggE, float* __restrict__ aggM,
                          float* __restrict__ counts, int N) {
  int i = blockIdx.x * 256 + threadIdx.x;
  if (i < N * 64) aggE[i] = 0.0f;
  if (i < N * 3)  aggM[i] = 0.0f;
  if (i < N)      counts[i] = 0.0f;
}

__global__ void egnn_degree(const int* __restrict__ recv, float* __restrict__ counts, int E) {
  int i = blockIdx.x * 256 + threadIdx.x;
  if (i < E) atomicAdd(&counts[recv[i]], 1.0f);
}

__global__ void egnn_init_nodes(const float* __restrict__ inputs,
                                const float* __restrict__ emb_w, const float* __restrict__ emb_b,
                                float* __restrict__ coord, float* __restrict__ vel,
                                float* __restrict__ h, unsigned short* __restrict__ h16u, int N) {
  int n = blockIdx.x * blockDim.x + threadIdx.x;
  if (n >= N) return;
  float px = inputs[n * 6 + 0], py = inputs[n * 6 + 1], pz = inputs[n * 6 + 2];
  float vx = inputs[n * 6 + 3], vy = inputs[n * 6 + 4], vz = inputs[n * 6 + 5];
  coord[n * 3 + 0] = px; coord[n * 3 + 1] = py; coord[n * 3 + 2] = pz;
  vel[n * 3 + 0] = vx; vel[n * 3 + 1] = vy; vel[n * 3 + 2] = vz;
  float s = sqrtf(vx * vx + vy * vy + vz * vz);
  _Float16* h16 = (_Float16*)h16u;
#pragma unroll 8
  for (int k = 0; k < 64; ++k) {
    float hv = s * emb_w[k] + emb_b[k];
    h[(size_t)n * 64 + k] = hv;
    h16[(size_t)n * 64 + k] = (_Float16)hv;
  }
}

__global__ void egnn_pack_out(const float* __restrict__ coord, const float* __restrict__ vel,
                              float* __restrict__ out, int N) {
  int n = blockIdx.x * blockDim.x + threadIdx.x;
  if (n >= N) return;
#pragma unroll
  for (int d = 0; d < 3; ++d) {
    out[n * 6 + d]     = coord[n * 3 + d];
    out[n * 6 + 3 + d] = vel[n * 3 + d];
  }
}

// ------------------------------------------------------------ edge kernel ---
// Per wave: 16-edge tiles. ef_in=[h_s|h_r|radial]; radial handled as rank-1
// update in the f32 accumulator. All matmuls via v_wmma_f32_16x16x32_f16.

__global__ __launch_bounds__(128) void egnn_edge_kernel(
    const float* __restrict__ coord, const unsigned short* __restrict__ h16u,
    const int* __restrict__ send, const int* __restrict__ recv,
    const float* __restrict__ W1, const float* __restrict__ B1,
    const float* __restrict__ W2, const float* __restrict__ B2,
    const float* __restrict__ CW1, const float* __restrict__ CB1,
    const float* __restrict__ CWO,
    float* __restrict__ aggE, float* __restrict__ aggM, int nE) {
  __shared__ __align__(32) u32 sW1s[16 * 256];   // 16 B-frags, fragment-major
  __shared__ __align__(32) u32 sW2s[8 * 256];    // 8 B-frags
  __shared__ __align__(32) u32 sCW1s[8 * 256];   // 8 B-frags
  __shared__ float sB1[64], sB2[64], sCB1[64], sW1L[64], sCW[64];
  __shared__ __align__(16) _Float16 sX1[WAVES][16 * 128];
  __shared__ __align__(16) _Float16 sX2[WAVES][16 * 64];
  __shared__ float sRad[WAVES][16];
  __shared__ float sCD[WAVES][48];
  __shared__ int sSend[WAVES][16];
  __shared__ int sRecv[WAVES][16];
  __shared__ float sCM[WAVES][16];

  const int tid = threadIdx.x;
  for (int i = tid; i < 4096; i += blockDim.x) {
    int pr, col, f;
    swz_idx(i, pr, col, f);
    sW1s[i] = pack2h(W1[(2 * pr) * 64 + col], W1[(2 * pr + 1) * 64 + col]);
    if (f < 8) {
      sW2s[i]  = pack2h(W2[(2 * pr) * 64 + col],  W2[(2 * pr + 1) * 64 + col]);
      sCW1s[i] = pack2h(CW1[(2 * pr) * 64 + col], CW1[(2 * pr + 1) * 64 + col]);
    }
  }
  for (int i = tid; i < 64; i += blockDim.x) {
    sB1[i] = B1[i]; sB2[i] = B2[i]; sCB1[i] = CB1[i];
    sW1L[i] = W1[128 * 64 + i];  // radial row of edge_w1
    sCW[i] = CWO[i];
  }
  __syncthreads();

  const int lane = tid & 31, wid = tid >> 5;
  const int hf = lane >> 4, cl = lane & 15;
  const uint4* hp4 = (const uint4*)h16u;  // h16 rows: [N][8] x uint4
  u32* x1u = (u32*)&sX1[wid][0];
  u32* x2u = (u32*)&sX2[wid][0];
  uint4* x1q = (uint4*)&sX1[wid][0];
  const int nT = (nE + 15) >> 4;

  for (int t0 = (int)blockIdx.x * WAVES + wid; t0 < nT; t0 += (int)gridDim.x * WAVES) {
    const int base = t0 << 4;
    if (lane < 16) {
      int e = base + lane; if (e >= nE) e = nE - 1;
      int s = send[e], r = recv[e];
      float dx = coord[3 * s + 0] - coord[3 * r + 0];
      float dy = coord[3 * s + 1] - coord[3 * r + 1];
      float dz = coord[3 * s + 2] - coord[3 * r + 2];
      sCD[wid][3 * lane + 0] = dx; sCD[wid][3 * lane + 1] = dy; sCD[wid][3 * lane + 2] = dz;
      sRad[wid][lane] = dx * dx + dy * dy + dz * dz;
      sSend[wid][lane] = s; sRecv[wid][lane] = r;
    }
    wave_fence();
    // gather X1 = [h16[send] | h16[recv]]  (16 rows x 16 uint4 = 256B/row)
#pragma unroll
    for (int i = lane; i < 256; i += 32) {
      int row = i >> 4, q = i & 15;
      int node = (q < 8) ? sSend[wid][row] : sRecv[wid][row];
      x1q[i] = hp4[node * 8 + (q & 7)];
    }
    wave_fence();
    // edge MLP layer 1: silu(X1 @ W1[0:128] + b1 + radial x W1[128])
#pragma unroll
    for (int tt = 0; tt < 4; ++tt) {
      const int colg = cl + 16 * tt;
      v8f acc;
#pragma unroll
      for (int j = 0; j < 8; ++j)
        acc[j] = sB1[colg] + sRad[wid][j + 8 * hf] * sW1L[colg];
#pragma unroll
      for (int c = 0; c < 4; ++c) {
        v16h a = load_afrag(x1u, 64, c, lane);
        v16h b = load_bfrag_swz(sW1s, c * 4 + tt, lane);
        acc = WMMA_F16(a, b, acc);
      }
#pragma unroll
      for (int j = 0; j < 8; ++j)
        sX2[wid][(j + 8 * hf) * 64 + colg] = (_Float16)silu_f(acc[j]);
    }
    wave_fence();
    // edge MLP layer 2 -> ef ; scatter-add ef to aggE[recv]
    _Float16* efb = &sX1[wid][0];  // reuse X1 scratch for ef (16x64 f16)
#pragma unroll
    for (int tt = 0; tt < 4; ++tt) {
      const int colg = cl + 16 * tt;
      v8f acc;
#pragma unroll
      for (int j = 0; j < 8; ++j) acc[j] = sB2[colg];
#pragma unroll
      for (int c = 0; c < 2; ++c) {
        v16h a = load_afrag(x2u, 32, c, lane);
        v16h b = load_bfrag_swz(sW2s, c * 4 + tt, lane);
        acc = WMMA_F16(a, b, acc);
      }
#pragma unroll
      for (int j = 0; j < 8; ++j) {
        const int m = j + 8 * hf;
        float v = silu_f(acc[j]);
        efb[m * 64 + colg] = (_Float16)v;
        if (base + m < nE)
          atomicAdd(&aggE[(size_t)sRecv[wid][m] * 64 + colg], v);
      }
    }
    wave_fence();
    // coord MLP: cm = silu(ef @ CW1 + cb1) @ cwout   (dot fused per lane)
    float part[8];
#pragma unroll
    for (int j = 0; j < 8; ++j) part[j] = 0.0f;
    const u32* efu = (const u32*)efb;
#pragma unroll
    for (int tt = 0; tt < 4; ++tt) {
      const int colg = cl + 16 * tt;
      v8f acc;
#pragma unroll
      for (int j = 0; j < 8; ++j) acc[j] = sCB1[colg];
#pragma unroll
      for (int c = 0; c < 2; ++c) {
        v16h a = load_afrag(efu, 32, c, lane);
        v16h b = load_bfrag_swz(sCW1s, c * 4 + tt, lane);
        acc = WMMA_F16(a, b, acc);
      }
#pragma unroll
      for (int j = 0; j < 8; ++j) part[j] += silu_f(acc[j]) * sCW[colg];
    }
#pragma unroll
    for (int mask = 1; mask < 16; mask <<= 1) {
#pragma unroll
      for (int j = 0; j < 8; ++j) part[j] += __shfl_xor(part[j], mask, 32);
    }
    if (cl == 0) {
#pragma unroll
      for (int j = 0; j < 8; ++j) sCM[wid][j + 8 * hf] = part[j];
    }
    wave_fence();
    // trans = clip(cd * cm); scatter-add to aggM[recv]
    if (lane < 16 && (base + lane) < nE) {
      const float cm = sCM[wid][lane];
      const int r = sRecv[wid][lane];
#pragma unroll
      for (int d = 0; d < 3; ++d) {
        float tr = sCD[wid][3 * lane + d] * cm;
        tr = fminf(fmaxf(tr, -100.0f), 100.0f);
        atomicAdd(&aggM[(size_t)r * 3 + d], tr);
      }
    }
    wave_fence();
  }
}

// ------------------------------------------------------------ node kernel ---

__global__ __launch_bounds__(128) void egnn_node_kernel(
    float* __restrict__ h, unsigned short* __restrict__ h16u,
    float* __restrict__ aggE, float* __restrict__ aggM,
    const float* __restrict__ counts,
    float* __restrict__ coord, float* __restrict__ vel,
    const float* __restrict__ NW1, const float* __restrict__ NB1,
    const float* __restrict__ NW2, const float* __restrict__ NB2,
    const float* __restrict__ VW1, const float* __restrict__ VB1,
    const float* __restrict__ VW2, const float* __restrict__ VB2,
    int N) {
  __shared__ __align__(32) u32 sNW1s[16 * 256];
  __shared__ __align__(32) u32 sNW2s[8 * 256];
  __shared__ __align__(32) u32 sVW1s[8 * 256];
  __shared__ float sNB1[64], sNB2[64], sVB1[64], sVW2c[64];
  __shared__ __align__(16) _Float16 sX1[WAVES][16 * 128];
  __shared__ __align__(16) _Float16 sX2[WAVES][16 * 64];
  __shared__ float sG[WAVES][16];

  const int tid = threadIdx.x;
  for (int i = tid; i < 4096; i += blockDim.x) {
    int pr, col, f;
    swz_idx(i, pr, col, f);
    sNW1s[i] = pack2h(NW1[(2 * pr) * 64 + col], NW1[(2 * pr + 1) * 64 + col]);
    if (f < 8) {
      sNW2s[i] = pack2h(NW2[(2 * pr) * 64 + col], NW2[(2 * pr + 1) * 64 + col]);
      sVW1s[i] = pack2h(VW1[(2 * pr) * 64 + col], VW1[(2 * pr + 1) * 64 + col]);
    }
  }
  for (int i = tid; i < 64; i += blockDim.x) {
    sNB1[i] = NB1[i]; sNB2[i] = NB2[i]; sVB1[i] = VB1[i]; sVW2c[i] = VW2[i];
  }
  __syncthreads();

  const float vb2 = VB2[0];
  const int lane = tid & 31, wid = tid >> 5;
  const int hf = lane >> 4, cl = lane & 15;
  _Float16* h16 = (_Float16*)h16u;
  const u32* hp = (const u32*)h16u;
  u32* x1u = (u32*)&sX1[wid][0];
  u32* x2u = (u32*)&sX2[wid][0];
  const int nT = (N + 15) >> 4;

  for (int t0 = (int)blockIdx.x * WAVES + wid; t0 < nT; t0 += (int)gridDim.x * WAVES) {
    const int base = t0 << 4;
    // X1 = [h | agg_sum] as f16; zero aggE after consuming (ready for next layer)
    for (int i = lane; i < 1024; i += 32) {
      int row = i >> 6, k2 = i & 63;
      int n = base + row; if (n >= N) n = N - 1;
      u32 d;
      if (k2 < 32) {
        d = hp[n * 32 + k2];
      } else {
        int kk = (k2 - 32) * 2;
        float a0 = aggE[(size_t)n * 64 + kk];
        float a1 = aggE[(size_t)n * 64 + kk + 1];
        if (base + row < N) {
          aggE[(size_t)n * 64 + kk] = 0.0f;
          aggE[(size_t)n * 64 + kk + 1] = 0.0f;
        }
        d = pack2h(a0, a1);
      }
      x1u[i] = d;
    }
    wave_fence();
    // node MLP layer 1
#pragma unroll
    for (int tt = 0; tt < 4; ++tt) {
      const int colg = cl + 16 * tt;
      v8f acc;
#pragma unroll
      for (int j = 0; j < 8; ++j) acc[j] = sNB1[colg];
#pragma unroll
      for (int c = 0; c < 4; ++c) {
        v16h a = load_afrag(x1u, 64, c, lane);
        v16h b = load_bfrag_swz(sNW1s, c * 4 + tt, lane);
        acc = WMMA_F16(a, b, acc);
      }
#pragma unroll
      for (int j = 0; j < 8; ++j)
        sX2[wid][(j + 8 * hf) * 64 + colg] = (_Float16)silu_f(acc[j]);
    }
    wave_fence();
    // node MLP layer 2 + residual -> h (f32) and h16
#pragma unroll
    for (int tt = 0; tt < 4; ++tt) {
      const int colg = cl + 16 * tt;
      v8f acc;
#pragma unroll
      for (int j = 0; j < 8; ++j) acc[j] = sNB2[colg];
#pragma unroll
      for (int c = 0; c < 2; ++c) {
        v16h a = load_afrag(x2u, 32, c, lane);
        v16h b = load_bfrag_swz(sNW2s, c * 4 + tt, lane);
        acc = WMMA_F16(a, b, acc);
      }
#pragma unroll
      for (int j = 0; j < 8; ++j) {
        const int m = j + 8 * hf;
        const int n = base + m;
        if (n < N) {
          float o = acc[j] + h[(size_t)n * 64 + colg];
          h[(size_t)n * 64 + colg] = o;
          h16[(size_t)n * 64 + colg] = (_Float16)o;
        }
      }
    }
    // vel gate: (silu(h_old @ VW1 + vb1) @ VW2 + vb2), h_old = X1 cols 0..63
    float part[8];
#pragma unroll
    for (int j = 0; j < 8; ++j) part[j] = 0.0f;
#pragma unroll
    for (int tt = 0; tt < 4; ++tt) {
      const int colg = cl + 16 * tt;
      v8f acc;
#pragma unroll
      for (int j = 0; j < 8; ++j) acc[j] = sVB1[colg];
#pragma unroll
      for (int c = 0; c < 2; ++c) {
        v16h a = load_afrag(x1u, 64, c, lane);  // chunks 0,1 = h part
        v16h b = load_bfrag_swz(sVW1s, c * 4 + tt, lane);
        acc = WMMA_F16(a, b, acc);
      }
#pragma unroll
      for (int j = 0; j < 8; ++j) part[j] += silu_f(acc[j]) * sVW2c[colg];
    }
#pragma unroll
    for (int mask = 1; mask < 16; mask <<= 1) {
#pragma unroll
      for (int j = 0; j < 8; ++j) part[j] += __shfl_xor(part[j], mask, 32);
    }
    if (cl == 0) {
#pragma unroll
      for (int j = 0; j < 8; ++j) sG[wid][j + 8 * hf] = part[j] + vb2;
    }
    wave_fence();
    // vel/coord update; zero aggM for next layer
    if (lane < 16 && (base + lane) < N) {
      const int n = base + lane;
      const float inv = 1.0f / fmaxf(counts[n], 1.0f);
      const float g = sG[wid][lane];
#pragma unroll
      for (int d = 0; d < 3; ++d) {
        float am = aggM[(size_t)n * 3 + d];
        aggM[(size_t)n * 3 + d] = 0.0f;
        float vo = vel[(size_t)n * 3 + d];
        float vn = am * inv + g * vo;
        vel[(size_t)n * 3 + d] = vn;
        coord[(size_t)n * 3 + d] += vn;
      }
    }
    wave_fence();
  }
}

// ------------------------------------------------------------------ host ---

extern "C" void kernel_launch(void* const* d_in, const int* in_sizes, int n_in,
                              void* d_out, int out_size, void* d_ws, size_t ws_size,
                              hipStream_t stream) {
  (void)n_in; (void)out_size; (void)ws_size;
  const int N = in_sizes[0] / 6;
  const int E = in_sizes[2];
  const int L = 4;

  const float* inputs = (const float*)d_in[0];
  const int* send = (const int*)d_in[2];
  const int* recv = (const int*)d_in[3];
  const float* emb_w = (const float*)d_in[4];
  const float* emb_b = (const float*)d_in[5];
  const float* ew1 = (const float*)d_in[6];
  const float* eb1 = (const float*)d_in[7];
  const float* ew2 = (const float*)d_in[8];
  const float* eb2 = (const float*)d_in[9];
  const float* nw1 = (const float*)d_in[10];
  const float* nb1 = (const float*)d_in[11];
  const float* nw2 = (const float*)d_in[12];
  const float* nb2 = (const float*)d_in[13];
  const float* cw1 = (const float*)d_in[14];
  const float* cb1 = (const float*)d_in[15];
  const float* cwo = (const float*)d_in[16];
  const float* vw1 = (const float*)d_in[17];
  const float* vb1 = (const float*)d_in[18];
  const float* vw2 = (const float*)d_in[19];
  const float* vb2 = (const float*)d_in[20];

  // carve workspace (256B aligned segments)
  size_t off = 0;
  char* base = (char*)d_ws;
  auto carve = [&](size_t bytes) -> void* {
    off = (off + 255) & ~(size_t)255;
    void* r = base + off;
    off += bytes;
    return r;
  };
  float* h            = (float*)carve((size_t)N * 64 * 4);
  unsigned short* h16 = (unsigned short*)carve((size_t)N * 64 * 2);
  float* coord        = (float*)carve((size_t)N * 3 * 4);
  float* vel          = (float*)carve((size_t)N * 3 * 4);
  float* aggE         = (float*)carve((size_t)N * 64 * 4);
  float* aggM         = (float*)carve((size_t)N * 3 * 4);
  float* cnts         = (float*)carve((size_t)N * 4);

  egnn_zero<<<(N * 64 + 255) / 256, 256, 0, stream>>>(aggE, aggM, cnts, N);
  egnn_degree<<<(E + 255) / 256, 256, 0, stream>>>(recv, cnts, E);
  egnn_init_nodes<<<(N + 127) / 128, 128, 0, stream>>>(inputs, emb_w, emb_b,
                                                       coord, vel, h, h16, N);

  const int nTe = (E + 15) / 16;
  int ebl = (nTe + WAVES - 1) / WAVES; if (ebl > 1920) ebl = 1920;
  const int nTn = (N + 15) / 16;
  int nbl = (nTn + WAVES - 1) / WAVES; if (nbl > 640) nbl = 640;

  for (int l = 0; l < L; ++l) {
    egnn_edge_kernel<<<ebl, 128, 0, stream>>>(
        coord, h16, send, recv,
        ew1 + (size_t)l * 129 * 64, eb1 + (size_t)l * 64,
        ew2 + (size_t)l * 64 * 64,  eb2 + (size_t)l * 64,
        cw1 + (size_t)l * 64 * 64,  cb1 + (size_t)l * 64,
        cwo + (size_t)l * 64,
        aggE, aggM, E);
    egnn_node_kernel<<<nbl, 128, 0, stream>>>(
        h, h16, aggE, aggM, cnts, coord, vel,
        nw1 + (size_t)l * 128 * 64, nb1 + (size_t)l * 64,
        nw2 + (size_t)l * 64 * 64,  nb2 + (size_t)l * 64,
        vw1 + (size_t)l * 64 * 64,  vb1 + (size_t)l * 64,
        vw2 + (size_t)l * 64,       vb2 + (size_t)l,
        N);
  }

  egnn_pack_out<<<(N + 127) / 128, 128, 0, stream>>>(coord, vel, (float*)d_out, N);
}
